// ColbertAdapter_59150289600537
// MI455X (gfx1250) — matrix operand order repaired and checked by reading, not compiled, not run on hardware
//
#include <hip/hip_runtime.h>

// ColBERT adapter for gfx1250 (MI455X): all contractions on v_wmma_f32_16x16x32_f16.
// Each wave carries TWO 16-row A fragments (32-row tile) so every B fragment
// (the expensive operand) feeds two WMMAs; block-shared operands are staged in
// LDS (single-barrier panels / double-buffered chunks) and read via ds_load_b128.
//
// Pipeline: [transpose weights->f16 N-major] -> [LN->f16 x3] -> [q/k/v GEMMs (LDS-A)]
//   -> [scores + MaxSim(U=4), LDS-staged K strip] -> [softmax rows]
//   -> [attn@V, double-buffered LDS V chunks] -> [out-proj GEMM] -> [LN4] -> [final GEMM]

typedef __attribute__((ext_vector_type(16))) _Float16 v16h;
typedef __attribute__((ext_vector_type(8)))  _Float16 v8h;
typedef __attribute__((ext_vector_type(8)))  float    v8f;

union Frag16 { v16h v; _Float16 h[16]; };

// A fragment (16x32, row-major, lda in halves).
// lanes 0-15: M=lane, K halves {k0..k0+7, k0+16..k0+23}
// lanes 16-31: M=lane-16, K halves {k0+8..k0+15, k0+24..k0+31}
static __device__ __forceinline__
v16h load_frag_a(const _Float16* __restrict__ A, int lda, int row0, int k0, int lane) {
    int r  = row0 + (lane & 15);
    int kb = k0 + ((lane >> 4) << 3);
    const _Float16* p = A + (size_t)r * lda + kb;
    Frag16 f;
    *(v8h*)&f.h[0] = *(const v8h*)(p);
    *(v8h*)&f.h[8] = *(const v8h*)(p + 16);
    return f.v;
}

// B fragment (32x16) from an N-major operand Bt[N][K] (ldb in halves).
// lane = N; lanes 0-15 hold K {k0..k0+7, k0+16..k0+23}, lanes 16-31 hold K {k0+8..k0+15, k0+24..k0+31}.
static __device__ __forceinline__
v16h load_frag_b(const _Float16* __restrict__ Bt, int ldb, int n0, int k0, int lane) {
    int n  = n0 + (lane & 15);
    int kb = k0 + ((lane >> 4) << 3);
    const _Float16* p = Bt + (size_t)n * ldb + kb;
    Frag16 f;
    *(v8h*)&f.h[0] = *(const v8h*)(p);
    *(v8h*)&f.h[8] = *(const v8h*)(p + 16);
    return f.v;
}

static __device__ __forceinline__
v8f wmma_f16(v16h a, v16h b, v8f c) {
    return __builtin_amdgcn_wmma_f32_16x16x32_f16(false, a, false, b, (short)0, c, false, false);
}

// ---------------------------------------------------------------------------
// Weight prep: f32 [K=512][N=512] -> f16 N-major wt[N=512][K=512]
__global__ __launch_bounds__(256) void k_transpose_w_f16(
    const float* __restrict__ w, _Float16* __restrict__ wt)
{
    int idx = blockIdx.x * 256 + threadIdx.x;    // grid = 1024 blocks
    int kk = idx >> 9, n = idx & 511;
    wt[(size_t)n * 512 + kk] = (_Float16)w[(size_t)kk * 512 + n];
}

// ---------------------------------------------------------------------------
// LayerNorm over D=512, one wave per row, output f16.
__global__ __launch_bounds__(256) void k_ln_rows_f16(
    const float* __restrict__ x, const float* __restrict__ w,
    const float* __restrict__ b, _Float16* __restrict__ y, int rows)
{
    const int lane = threadIdx.x & 31, wave = threadIdx.x >> 5;
    int row = blockIdx.x * 8 + wave;
    if (row >= rows) return;
    const float* xr = x + (size_t)row * 512;
    float v[16], s = 0.f, sq = 0.f;
#pragma unroll
    for (int i = 0; i < 16; ++i) {
        float t = xr[lane + 32 * i];
        v[i] = t; s += t; sq += t * t;
    }
#pragma unroll
    for (int m = 16; m >= 1; m >>= 1) {
        s  += __shfl_xor(s,  m, 32);
        sq += __shfl_xor(sq, m, 32);
    }
    float mu  = s * (1.f / 512.f);
    float var = sq * (1.f / 512.f) - mu * mu;
    float rs  = rsqrtf(var + 1e-5f);
    _Float16* yr = y + (size_t)row * 512;
#pragma unroll
    for (int i = 0; i < 16; ++i) {
        int c = lane + 32 * i;
        yr[c] = (_Float16)((v[i] - mu) * rs * w[c] + b[c]);
    }
}

// ---------------------------------------------------------------------------
// GEMM with K == 512: A[M][512] f16 row-major x Bt[N=512][512] f16 N-major.
// Block = 32-row M tile x all of N. The 32x512 A panel (32 KB) is staged to
// LDS once (single barrier); each of the 8 waves owns a 64-wide N strip,
// holds two A fragments + 8 accumulators, and reuses each global B fragment
// for two WMMAs.
// mode 0: f16 out[M][N]  mode 1: f16 out[N][M] (transposed)  mode 2: f32 out[M][N]
#define GEMM_K   512
#define A_LDS_LD 520   // 512 + 8 halves pad: 16B-aligned rows, staggers LDS banks
__global__ __launch_bounds__(256) void k_gemm_wmma(
    const _Float16* __restrict__ A, const _Float16* __restrict__ Bt,
    const float* __restrict__ bias, void* __restrict__ outp,
    int M, int N, float alpha, int mode)
{
    __shared__ __align__(16) _Float16 As[32 * A_LDS_LD];
    const int lane = threadIdx.x & 31, wave = threadIdx.x >> 5;
    const int row0 = blockIdx.x << 5;

    // Cooperative stage: 8 threads per row, 64 halves (128 B) each.
    {
        int r  = threadIdx.x >> 3;
        int c0 = (threadIdx.x & 7) << 6;
        const _Float16* src = A + (size_t)(row0 + r) * GEMM_K + c0;
        _Float16* dst = As + r * A_LDS_LD + c0;
#pragma unroll
        for (int j = 0; j < 8; ++j)
            *(v8h*)(dst + j * 8) = *(const v8h*)(src + j * 8);
    }
    __syncthreads();

    const int n0 = wave << 6;                    // 8 waves x 64-col strips = N=512
    v8f acc[2][4] = {};
    for (int k0 = 0; k0 < GEMM_K; k0 += 32) {
        v16h a0 = load_frag_a(As, A_LDS_LD, 0,  k0, lane);   // ds_load_b128 x2
        v16h a1 = load_frag_a(As, A_LDS_LD, 16, k0, lane);
#pragma unroll
        for (int s = 0; s < 4; ++s) {
            v16h b = load_frag_b(Bt, GEMM_K, n0 + s * 16, k0, lane);
            acc[0][s] = wmma_f16(a0, b, acc[0][s]);
            acc[1][s] = wmma_f16(a1, b, acc[1][s]);
        }
    }
    const int nl = lane & 15, hl = lane >> 4;
#pragma unroll
    for (int s = 0; s < 4; ++s) {
        int n = n0 + s * 16 + nl;
        float bs = bias ? bias[n] : 0.f;
#pragma unroll
        for (int half = 0; half < 2; ++half) {
#pragma unroll
            for (int r = 0; r < 8; ++r) {
                int m = row0 + half * 16 + r + hl * 8;
                float val = (acc[half][s][r] + bs) * alpha;
                if (mode == 0)      ((_Float16*)outp)[(size_t)m * N + n] = (_Float16)val;
                else if (mode == 1) ((_Float16*)outp)[(size_t)n * M + m] = (_Float16)val;
                else                ((float*)outp)[(size_t)m * N + n] = val;
            }
        }
    }
}

// ---------------------------------------------------------------------------
// ColBERT scores + MaxSim. q pre-scaled by 1/sqrt(dk) in its GEMM epilogue.
// Block = (bh, C-tile): the 64x64 K strip for this head/C-tile is staged to
// LDS once and shared by 8 waves; each wave computes TWO 16(T) x 64 pre-max
// tiles (T-tiles wave and wave+8), so each LDS B fragment feeds two WMMAs.
#define KS_LD 72   // 64 + 8 halves pad
__global__ __launch_bounds__(256) void k_colbert_scores(
    const _Float16* __restrict__ q,   // [B*T][512], head h at cols h*64..
    const _Float16* __restrict__ k,   // [C*U][512]
    float* __restrict__ scores)       // [B*H][256][2048]
{
    __shared__ __align__(16) _Float16 Ks[64 * KS_LD];
    const int lane = threadIdx.x & 31, wave = threadIdx.x >> 5;
    int bid = blockIdx.x;                        // 32 * 128 = 4096 blocks
    int bh  = bid >> 7;
    int tC  = bid & 127;
    int b = bh >> 3, h = bh & 7;
    int n0 = tC << 6;                            // pre-max col base = c*4 + u

    // Stage K strip [64 rows x 64 halves]: 4 threads per row, 16 halves each.
    {
        int r  = threadIdx.x >> 2;
        int c0 = (threadIdx.x & 3) << 4;
        const _Float16* src = k + (size_t)(n0 + r) * 512 + h * 64 + c0;
        _Float16* dst = Ks + r * KS_LD + c0;
        *(v8h*)(dst)     = *(const v8h*)(src);
        *(v8h*)(dst + 8) = *(const v8h*)(src + 8);
    }
    __syncthreads();

    const _Float16* qh = q + (size_t)(b * 256) * 512 + h * 64;
    int rowA = wave << 4;                        // T-tile `wave`
    int rowB = (wave + 8) << 4;                  // T-tile `wave+8`

    v8f acc[2][4] = {};
#pragma unroll
    for (int kc = 0; kc < 64; kc += 32) {
        v16h a0 = load_frag_a(qh, 512, rowA, kc, lane);
        v16h a1 = load_frag_a(qh, 512, rowB, kc, lane);
#pragma unroll
        for (int s = 0; s < 4; ++s) {
            v16h bb = load_frag_b(Ks, KS_LD, s * 16, kc, lane);   // from LDS
            acc[0][s] = wmma_f16(a0, bb, acc[0][s]);
            acc[1][s] = wmma_f16(a1, bb, acc[1][s]);
        }
    }
    const int nl = lane & 15, hl = lane >> 4;
    float* sb = scores + (size_t)bh * 256 * 2048;
#pragma unroll
    for (int half = 0; half < 2; ++half) {
        int rbase = (half ? rowB : rowA);
#pragma unroll
        for (int s = 0; s < 4; ++s) {
#pragma unroll
            for (int r = 0; r < 8; ++r) {
                float x = acc[half][s][r];       // max over U=4 (fastest N index)
                x = fmaxf(x, __shfl_xor(x, 1, 32));
                x = fmaxf(x, __shfl_xor(x, 2, 32));
                if ((nl & 3) == 0) {
                    int t = rbase + r + hl * 8;
                    int c = (tC << 4) + (s << 2) + (nl >> 2);
                    sb[(size_t)t * 2048 + c] = x;
                }
            }
        }
    }
}

// ---------------------------------------------------------------------------
// Softmax over C=2048, one block per row; output attn f16.
__global__ __launch_bounds__(256) void k_softmax_rows(
    const float* __restrict__ scores, _Float16* __restrict__ attn)
{
    __shared__ float red[256];
    int tid = threadIdx.x;
    const float* sr = scores + (size_t)blockIdx.x * 2048;
    float v[8], mx = -1e30f;
#pragma unroll
    for (int i = 0; i < 8; ++i) { v[i] = sr[tid + i * 256]; mx = fmaxf(mx, v[i]); }
    red[tid] = mx; __syncthreads();
    for (int s = 128; s > 0; s >>= 1) {
        if (tid < s) red[tid] = fmaxf(red[tid], red[tid + s]);
        __syncthreads();
    }
    mx = red[0]; __syncthreads();
    float sum = 0.f;
#pragma unroll
    for (int i = 0; i < 8; ++i) { v[i] = __expf(v[i] - mx); sum += v[i]; }
    red[tid] = sum; __syncthreads();
    for (int s = 128; s > 0; s >>= 1) {
        if (tid < s) red[tid] += red[tid + s];
        __syncthreads();
    }
    float inv = 1.f / red[0];
    _Float16* ar = attn + (size_t)blockIdx.x * 2048;
#pragma unroll
    for (int i = 0; i < 8; ++i) ar[tid + i * 256] = (_Float16)(v[i] * inv);
}

// ---------------------------------------------------------------------------
// ctx = attn @ V_h : attn [B*H][256][2048] f16, vT [512][2048] f16 (N-major).
// One block per (b,h): 8 waves x 2 T-tiles cover T=256. The shared V strip is
// double-buffered in LDS in 64x32 chunks (stage next while computing current,
// one barrier per step); each LDS B fragment feeds two WMMAs.
#define VS_LD 40   // 32 + 8 halves pad
__global__ __launch_bounds__(256) void k_attn_v(
    const _Float16* __restrict__ attn, const _Float16* __restrict__ vT,
    _Float16* __restrict__ ctx)
{
    __shared__ __align__(16) _Float16 Vs[2][64 * VS_LD];
    const int lane = threadIdx.x & 31, wave = threadIdx.x >> 5;
    int bh = blockIdx.x;                         // 32 blocks
    int b = bh >> 3, h = bh & 7;
    const _Float16* ah = attn + (size_t)bh * 256 * 2048;
    const _Float16* vh = vT + (size_t)(h * 64) * 2048;
    int rowA = wave << 4;
    int rowB = (wave + 8) << 4;

    // Per-chunk stage: 4 threads per row, 8 halves (16 B) each.
    const int sr = threadIdx.x >> 2;
    const int sc = (threadIdx.x & 3) << 3;
    const _Float16* ssrc = vh + (size_t)sr * 2048 + sc;
    _Float16* sdst = &Vs[0][0] + sr * VS_LD + sc;

    *(v8h*)(sdst) = *(const v8h*)(ssrc);         // stage chunk 0 into buf 0
    __syncthreads();

    v8f acc[2][4] = {};
    for (int kc = 0; kc < 2048; kc += 32) {
        int cur = (kc >> 5) & 1;
        if (kc + 32 < 2048)                       // prefetch next chunk
            *(v8h*)(sdst + (cur ^ 1) * 64 * VS_LD) = *(const v8h*)(ssrc + kc + 32);
        const _Float16* vb = &Vs[cur][0];
        v16h a0 = load_frag_a(ah, 2048, rowA, kc, lane);
        v16h a1 = load_frag_a(ah, 2048, rowB, kc, lane);
#pragma unroll
        for (int s = 0; s < 4; ++s) {
            v16h bb = load_frag_b(vb, VS_LD, s * 16, 0, lane);    // from LDS
            acc[0][s] = wmma_f16(a0, bb, acc[0][s]);
            acc[1][s] = wmma_f16(a1, bb, acc[1][s]);
        }
        __syncthreads();   // next buffer staged; current buffer free to overwrite
    }
    const int nl = lane & 15, hl = lane >> 4;
#pragma unroll
    for (int half = 0; half < 2; ++half) {
        int rbase = (half ? rowB : rowA);
#pragma unroll
        for (int s = 0; s < 4; ++s) {
            int n = s * 16 + nl;
#pragma unroll
            for (int r = 0; r < 8; ++r) {
                int m = rbase + r + hl * 8;
                ctx[(size_t)(b * 256 + m) * 512 + h * 64 + n] = (_Float16)acc[half][s][r];
            }
        }
    }
}

// ---------------------------------------------------------------------------
extern "C" void kernel_launch(void* const* d_in, const int* in_sizes, int n_in,
                              void* d_out, int out_size, void* d_ws, size_t ws_size,
                              hipStream_t stream)
{
    (void)in_sizes; (void)n_in; (void)out_size; (void)ws_size;
    const float* model = (const float*)d_in[0];
    const float* ckey  = (const float*)d_in[1];
    const float* cval  = (const float*)d_in[2];
    const float* ln1w = (const float*)d_in[3],  *ln1b = (const float*)d_in[4];
    const float* ln2w = (const float*)d_in[5],  *ln2b = (const float*)d_in[6];
    const float* ln3w = (const float*)d_in[7],  *ln3b = (const float*)d_in[8];
    const float* ln4w = (const float*)d_in[9],  *ln4b = (const float*)d_in[10];
    const float* wq = (const float*)d_in[11], *bq = (const float*)d_in[12];
    const float* wk = (const float*)d_in[13], *bk = (const float*)d_in[14];
    const float* wv = (const float*)d_in[15], *bv = (const float*)d_in[16];
    const float* wo = (const float*)d_in[17], *bo = (const float*)d_in[18];
    const float* wp = (const float*)d_in[19], *bp = (const float*)d_in[20];
    float* out = (float*)d_out;

    // Workspace carve-out (~126 MB total).
    char* ws = (char*)d_ws;
    size_t off = 0;
    auto carve = [&](size_t bytes) {
        void* p = ws + off;
        off = (off + bytes + 255) & ~(size_t)255;
        return p;
    };
    _Float16* wqt = (_Float16*)carve(512 * 512 * 2);
    _Float16* wkt = (_Float16*)carve(512 * 512 * 2);
    _Float16* wvt = (_Float16*)carve(512 * 512 * 2);
    _Float16* wot = (_Float16*)carve(512 * 512 * 2);
    _Float16* wpt = (_Float16*)carve(512 * 512 * 2);
    _Float16* qn  = (_Float16*)carve((size_t)1024 * 512 * 2);
    _Float16* kn  = (_Float16*)carve((size_t)8192 * 512 * 2);
    _Float16* vn  = (_Float16*)carve((size_t)2048 * 512 * 2);
    _Float16* qf  = (_Float16*)carve((size_t)1024 * 512 * 2);
    _Float16* kf  = (_Float16*)carve((size_t)8192 * 512 * 2);
    _Float16* vTf = (_Float16*)carve((size_t)512 * 2048 * 2);
    float*    sc  = (float*)   carve((size_t)32 * 256 * 2048 * 4);   // 64 MB
    _Float16* at  = (_Float16*)carve((size_t)32 * 256 * 2048 * 2);   // 32 MB
    _Float16* ctx = (_Float16*)carve((size_t)1024 * 512 * 2);
    float*    prj = (float*)   carve((size_t)1024 * 512 * 4);
    _Float16* pln = (_Float16*)carve((size_t)1024 * 512 * 2);

    const dim3 blk(256);

    // 1) weights -> f16 N-major
    k_transpose_w_f16<<<1024, blk, 0, stream>>>(wq, wqt);
    k_transpose_w_f16<<<1024, blk, 0, stream>>>(wk, wkt);
    k_transpose_w_f16<<<1024, blk, 0, stream>>>(wv, wvt);
    k_transpose_w_f16<<<1024, blk, 0, stream>>>(wo, wot);
    k_transpose_w_f16<<<1024, blk, 0, stream>>>(wp, wpt);

    // 2) pre-attention LayerNorms -> f16
    k_ln_rows_f16<<<1024 / 8, blk, 0, stream>>>(model, ln1w, ln1b, qn, 1024);
    k_ln_rows_f16<<<8192 / 8, blk, 0, stream>>>(ckey,  ln2w, ln2b, kn, 8192);
    k_ln_rows_f16<<<2048 / 8, blk, 0, stream>>>(cval,  ln3w, ln3b, vn, 2048);

    // 3) q/k/v projections (q folds 1/sqrt(64); v stored transposed for B use)
    //    grid = M/32 blocks (one 32-row A panel per block)
    k_gemm_wmma<<<1024 / 32, blk, 0, stream>>>(qn, wqt, bq, qf, 1024, 512, 0.125f, 0);
    k_gemm_wmma<<<8192 / 32, blk, 0, stream>>>(kn, wkt, bk, kf, 8192, 512, 1.0f,   0);
    k_gemm_wmma<<<2048 / 32, blk, 0, stream>>>(vn, wvt, bv, vTf, 2048, 512, 1.0f,  1);

    // 4) ColBERT MaxSim scores: 32 bh * 128 C-tiles blocks (block = all 16 T-tiles)
    k_colbert_scores<<<32 * 128, blk, 0, stream>>>(qf, kf, sc);

    // 5) softmax over C per (b,h,t) row
    k_softmax_rows<<<32 * 256, blk, 0, stream>>>(sc, at);

    // 6) attn @ V -> ctx f16 [1024][512]: one block per (b,h)
    k_attn_v<<<32, blk, 0, stream>>>(at, vTf, ctx);

    // 7) output projection (f32), LN4 (f16), final projection -> d_out (f32)
    k_gemm_wmma<<<1024 / 32, blk, 0, stream>>>(ctx, wot, bo, prj, 1024, 512, 1.0f, 2);
    k_ln_rows_f16<<<1024 / 8, blk, 0, stream>>>(prj, ln4w, ln4b, pln, 1024);
    k_gemm_wmma<<<1024 / 32, blk, 0, stream>>>(pln, wpt, bp, out, 1024, 512, 1.0f, 2);
}
